// BaseAttentionProductWithLSE_72121090834674
// MI455X (gfx1250) — compile-verified, matrix operand order checked by reading, not run on hardware
//
#include <hip/hip_runtime.h>
#include <math.h>

typedef __attribute__((ext_vector_type(16))) _Float16 v16h;
typedef __attribute__((ext_vector_type(8)))  _Float16 v8h;
typedef __attribute__((ext_vector_type(4)))  _Float16 v4h;
typedef __attribute__((ext_vector_type(8)))  float    v8f;
typedef __attribute__((ext_vector_type(4)))  float    v4f;

#define HH 16
#define SS 4096
#define DD 64
#define KV_TILE 32
#define NCHUNK (SS / KV_TILE)
#define WAVES 4
#define QROWS_PER_BLOCK (WAVES * 16)

#define INVLN2 1.44269504088896340736f
#define LN2    0.69314718055994530942f

// CDNA5 async copy: HBM -> LDS, no VGPR round trip, tracked by ASYNCcnt.
__device__ __forceinline__ void async_load_b128_to_lds(uint32_t lds_off, const void* gp) {
    uint64_t ga = (uint64_t)(uintptr_t)gp;
    asm volatile("global_load_async_to_lds_b128 %0, %1, off"
                 :: "v"(lds_off), "v"(ga) : "memory");
}
__device__ __forceinline__ void wait_asynccnt0() {
    asm volatile("s_wait_asynccnt 0x0" ::: "memory");
}

// Fused DPP max step: one VOP2 computing max(dpp(x), x). DPP applies to src0.
#define DPP_MAX_STEP(x, CTRL)                                              \
    do {                                                                   \
        float _r;                                                          \
        asm("v_max_num_f32_dpp %0, %1, %1 " CTRL                           \
            " row_mask:0xf bank_mask:0xf bound_ctrl:1"                     \
            : "=v"(_r) : "v"(x));                                          \
        (x) = _r;                                                          \
    } while (0)

// Full 16-lane row max (stays inside the 16-lane DPP row = one C-layout half):
// xor1, xor2, then quad-combine (xor7) and oct-combine (xor15) on uniform groups.
__device__ __forceinline__ float row_max16(float x) {
    DPP_MAX_STEP(x, "quad_perm:[1,0,3,2]");   // xor 1
    DPP_MAX_STEP(x, "quad_perm:[2,3,0,1]");   // xor 2
    DPP_MAX_STEP(x, "row_half_mirror");       // xor 7 (quads uniform)
    DPP_MAX_STEP(x, "row_mirror");            // xor 15 (octs uniform)
    return x;
}

// FlashAttention fwd, fp32 in/out, f16 WMMA compute / f32 accumulate.
// One wave = one 16-row Q tile; 4 waves share double-buffered f16 K/V LDS tiles.
// K/V stream HBM->LDS with async loads; each thread converts the slots its own
// async loads filled (f32 -> f16, V transposed) after s_wait_asynccnt.
// Softmax runs in the exp2 domain; row-max via fused DPP16 max; row-sum via
// an all-ones-B WMMA (C layout broadcasts ell across the half — no shuffles).
__global__ __launch_bounds__(128)
void flash_attn_fwd_lse(const float* __restrict__ q,
                        const float* __restrict__ k,
                        const float* __restrict__ v,
                        const float* __restrict__ mask,
                        float* __restrict__ out_ctx,
                        float* __restrict__ out_lse)
{
    __shared__ alignas(16) float    kRaw[KV_TILE * DD];         // async landing pad (f32)
    __shared__ alignas(16) float    vRaw[KV_TILE * DD];         // async landing pad (f32)
    __shared__ alignas(16) _Float16 k16[2][KV_TILE * DD];       // f16 [key][dim]
    __shared__ alignas(16) _Float16 vT16[2][DD * KV_TILE];      // f16 [dim][key] (transposed)
    __shared__ alignas(16) _Float16 pbuf[WAVES][16 * KV_TILE];  // per-wave P staging (f16)

    const int tid  = threadIdx.x;
    const int wave = tid >> 5;
    const int lane = tid & 31;
    const int half = lane >> 4;   // 0: lanes 0-15, 1: lanes 16-31
    const int l16  = lane & 15;

    const int h     = blockIdx.y;
    const int qbase = blockIdx.x * QROWS_PER_BLOCK + wave * 16;

    const float scale2 = 0.125f * INVLN2;   // 1/sqrt(64) folded with 1/ln2 (exp2 domain)

    const size_t headOff = (size_t)h * SS * DD;
    const float* kGlob = k + headOff;
    const float* vGlob = v + headOff;

    // ---- Q tile (16x64) -> two f16 A fragments (contraction dims 0..31 / 32..63) ----
    // 16-bit A layout: lane = M row; element e -> K = e + (e>=8 ? 8 : 0) + half*8 (+32 frag 1)
    v16h aq0, aq1;
    {
        const float* qp = q + ((size_t)(h * SS + qbase + l16)) * DD + half * 8;
        #pragma unroll
        for (int j = 0; j < 4; ++j) {
            float2 a0 = *(const float2*)(qp + 2 * j);
            float2 a1 = *(const float2*)(qp + 16 + 2 * j);
            float2 b0 = *(const float2*)(qp + 32 + 2 * j);
            float2 b1 = *(const float2*)(qp + 48 + 2 * j);
            aq0[2*j]   = (_Float16)(a0.x * scale2); aq0[2*j+1]   = (_Float16)(a0.y * scale2);
            aq0[8+2*j] = (_Float16)(a1.x * scale2); aq0[8+2*j+1] = (_Float16)(a1.y * scale2);
            aq1[2*j]   = (_Float16)(b0.x * scale2); aq1[2*j+1]   = (_Float16)(b0.y * scale2);
            aq1[8+2*j] = (_Float16)(b1.x * scale2); aq1[8+2*j+1] = (_Float16)(b1.y * scale2);
        }
    }

    // All-ones B matrix: rowsum(P) via WMMA.
    v16h bones;
    #pragma unroll
    for (int e = 0; e < 16; ++e) bones[e] = (_Float16)1.0f;

    // Online-softmax state (log2 units). C layout: VGPR i -> row m = i + 8*half.
    float mrow[8];
    v8f o[5] = {v8f{}, v8f{}, v8f{}, v8f{}, v8f{}};  // 4 context tiles + rowsum tile
    #pragma unroll
    for (int i = 0; i < 8; ++i) mrow[i] = -1e30f;

    // Per-thread staging geometry: 4 float4 slots (idx = tid + 128*r).
    // row = key (0..31), colF = float4 column within the 64-dim row.

    // ---- prologue: async-stage chunk 0, convert into buffer 0 ----
    #pragma unroll
    for (int r = 0; r < 4; ++r) {
        const int idx = tid + 128 * r;
        const int row = idx >> 4, colF = idx & 15;
        async_load_b128_to_lds((uint32_t)(uintptr_t)&kRaw[idx * 4],
                               kGlob + (size_t)row * DD + colF * 4);
        async_load_b128_to_lds((uint32_t)(uintptr_t)&vRaw[idx * 4],
                               vGlob + (size_t)row * DD + colF * 4);
    }
    wait_asynccnt0();
    #pragma unroll
    for (int r = 0; r < 4; ++r) {
        const int idx = tid + 128 * r;
        const int row = idx >> 4, colF = idx & 15;
        v4f kx = *(const v4f*)&kRaw[idx * 4];
        v4h kh = { (_Float16)kx.x, (_Float16)kx.y, (_Float16)kx.z, (_Float16)kx.w };
        *(v4h*)&k16[0][idx * 4] = kh;
        v4f vx = *(const v4f*)&vRaw[idx * 4];
        vT16[0][(colF*4 + 0) * KV_TILE + row] = (_Float16)vx.x;
        vT16[0][(colF*4 + 1) * KV_TILE + row] = (_Float16)vx.y;
        vT16[0][(colF*4 + 2) * KV_TILE + row] = (_Float16)vx.z;
        vT16[0][(colF*4 + 3) * KV_TILE + row] = (_Float16)vx.w;
    }
    __syncthreads();

    for (int c = 0; c < NCHUNK; ++c) {
        const int p = c & 1;
        const int kbase = c * KV_TILE;
        const bool more = (c + 1 < NCHUNK);

        // ---- kick off chunk c+1 HBM->LDS async (overlaps with compute) ----
        if (more) {
            const int nb = kbase + KV_TILE;
            #pragma unroll
            for (int r = 0; r < 4; ++r) {
                const int idx = tid + 128 * r;
                const int row = idx >> 4, colF = idx & 15;
                async_load_b128_to_lds((uint32_t)(uintptr_t)&kRaw[idx * 4],
                                       kGlob + (size_t)(nb + row) * DD + colF * 4);
                async_load_b128_to_lds((uint32_t)(uintptr_t)&vRaw[idx * 4],
                                       vGlob + (size_t)(nb + row) * DD + colF * 4);
            }
        }

        // ---- scores: S(16x32) = Q . K^T, two 16x16 C tiles (pure f16 LDS loads) ----
        const _Float16* kT = k16[p];
        v8f c0 = {}, c1 = {};
        #pragma unroll
        for (int g = 0; g < 2; ++g) {          // key sub-tile (cols 16g..16g+15)
            v8f acc = {};
            #pragma unroll
            for (int cb = 0; cb < 2; ++cb) {   // contraction dims 32*cb..32*cb+31
                const _Float16* bp = &kT[(16 * g + l16) * DD + 32 * cb + half * 8];
                const v8h lo = *(const v8h*)(bp);
                const v8h hi = *(const v8h*)(bp + 16);
                v16h bk;
                #pragma unroll
                for (int e = 0; e < 8; ++e) { bk[e] = lo[e]; bk[8 + e] = hi[e]; }
                acc = __builtin_amdgcn_wmma_f32_16x16x32_f16(
                          false, (cb ? aq1 : aq0), false, bk, (short)0, acc, false, false);
            }
            if (g == 0) c0 = acc; else c1 = acc;
        }

        // ---- online softmax in exp2 domain (max via DPP16; sum via ones-WMMA) ----
        const float mk0 = mask[kbase + l16] * INVLN2;
        const float mk1 = mask[kbase + 16 + l16] * INVLN2;
        float p0[8], p1[8];
        v8f av;
        #pragma unroll
        for (int i = 0; i < 8; ++i) {
            const float s0 = c0[i] + mk0;
            const float s1 = c1[i] + mk1;
            const float r  = row_max16(fmaxf(s0, s1));  // 4 fused DPP-max ops
            const float mn = fmaxf(mrow[i], r);
            av[i] = exp2f(mrow[i] - mn);
            p0[i] = exp2f(s0 - mn);
            p1[i] = exp2f(s1 - mn);
            mrow[i] = mn;
        }
        #pragma unroll
        for (int t = 0; t < 5; ++t)
            o[t] *= av;                         // packs into v_pk_mul_f32 pairs

        // ---- P: C layout -> row-major f16 LDS -> A fragment (K = 32 keys) ----
        #pragma unroll
        for (int i = 0; i < 8; ++i) {
            const int m = i + 8 * half;
            pbuf[wave][m * KV_TILE + l16]      = (_Float16)p0[i];
            pbuf[wave][m * KV_TILE + 16 + l16] = (_Float16)p1[i];
        }
        v16h ap;
        {
            const v8h lo = *(const v8h*)&pbuf[wave][l16 * KV_TILE + half * 8];
            const v8h hi = *(const v8h*)&pbuf[wave][l16 * KV_TILE + 16 + half * 8];
            #pragma unroll
            for (int e = 0; e < 8; ++e) { ap[e] = lo[e]; ap[8 + e] = hi[e]; }
        }

        // ---- O(16x64) += P . V (4 tiles), rowsum tile via all-ones B ----
        const _Float16* vT = vT16[p];
        #pragma unroll
        for (int t = 0; t < 4; ++t) {
            const v8h lo = *(const v8h*)&vT[(16 * t + l16) * KV_TILE + half * 8];
            const v8h hi = *(const v8h*)&vT[(16 * t + l16) * KV_TILE + 16 + half * 8];
            v16h bv;
            #pragma unroll
            for (int e = 0; e < 8; ++e) { bv[e] = lo[e]; bv[8 + e] = hi[e]; }
            o[t] = __builtin_amdgcn_wmma_f32_16x16x32_f16(
                       false, ap, false, bv, (short)0, o[t], false, false);
        }
        o[4] = __builtin_amdgcn_wmma_f32_16x16x32_f16(
                   false, ap, false, bones, (short)0, o[4], false, false);

        // ---- land chunk c+1: wait own async loads, convert own slots to f16 ----
        if (more) {
            wait_asynccnt0();
            #pragma unroll
            for (int r = 0; r < 4; ++r) {
                const int idx = tid + 128 * r;
                const int row = idx >> 4, colF = idx & 15;
                v4f kx = *(const v4f*)&kRaw[idx * 4];
                v4h kh = { (_Float16)kx.x, (_Float16)kx.y, (_Float16)kx.z, (_Float16)kx.w };
                *(v4h*)&k16[p ^ 1][idx * 4] = kh;
                v4f vx = *(const v4f*)&vRaw[idx * 4];
                vT16[p ^ 1][(colF*4 + 0) * KV_TILE + row] = (_Float16)vx.x;
                vT16[p ^ 1][(colF*4 + 1) * KV_TILE + row] = (_Float16)vx.y;
                vT16[p ^ 1][(colF*4 + 2) * KV_TILE + row] = (_Float16)vx.z;
                vT16[p ^ 1][(colF*4 + 3) * KV_TILE + row] = (_Float16)vx.w;
            }
        }
        __syncthreads();   // conversions visible; dscnt wait orders kRaw/vRaw reuse
    }

    // ---- epilogue: ell = rowsum tile (broadcast across half), normalize, write ----
    float invl[8], lrow[8];
    #pragma unroll
    for (int i = 0; i < 8; ++i) { lrow[i] = o[4][i]; invl[i] = 1.0f / lrow[i]; }

    #pragma unroll
    for (int t = 0; t < 4; ++t) {
        #pragma unroll
        for (int i = 0; i < 8; ++i) {
            const int row = qbase + i + 8 * half;
            out_ctx[((size_t)(h * SS) + row) * DD + 16 * t + l16] = o[t][i] * invl[i];
        }
    }
    if (l16 == 0) {   // lanes 0 / 16 hold rows 0..7 / 8..15
        #pragma unroll
        for (int i = 0; i < 8; ++i) {
            const int row = qbase + i + 8 * half;
            out_lse[(size_t)h * SS + row] = (mrow[i] + __log2f(lrow[i])) * LN2;
        }
    }
}

extern "C" void kernel_launch(void* const* d_in, const int* in_sizes, int n_in,
                              void* d_out, int out_size, void* d_ws, size_t ws_size,
                              hipStream_t stream) {
    const float* q    = (const float*)d_in[0];
    const float* k    = (const float*)d_in[1];
    const float* v    = (const float*)d_in[2];
    const float* mask = (const float*)d_in[3];

    float* out_ctx = (float*)d_out;                    // [1,16,4096,64]
    float* out_lse = out_ctx + (size_t)HH * SS * DD;   // [1,16,4096,1]

    dim3 grid(SS / QROWS_PER_BLOCK, HH, 1);
    dim3 block(128, 1, 1);
    flash_attn_fwd_lse<<<grid, block, 0, stream>>>(q, k, v, mask, out_ctx, out_lse);
}
